// FlowGraphSAGE_30339648979088
// MI455X (gfx1250) — compile-verified, matrix operand order checked by reading, not run on hardware
//
#include <hip/hip_runtime.h>

#define N_NODES 100000
#define N_EDGES 1600000
#define DIM 128

typedef __attribute__((ext_vector_type(16))) __bf16        v16bf;
typedef __attribute__((ext_vector_type(8)))  float         v8f;
typedef __attribute__((ext_vector_type(8)))  unsigned int  v8u;

// round-to-nearest-even f32 -> bf16 bit pattern (cold path: weight conversion)
__device__ __forceinline__ unsigned short f2bfu(float f) {
    unsigned u = __builtin_bit_cast(unsigned, f);
    unsigned r = u + 0x7FFFu + ((u >> 16) & 1u);
    return (unsigned short)(r >> 16);
}

// hot path: pack two f32 into two bf16 (truncate) with one v_perm_b32
// result = { hi[31:16], lo[31:16] }
__device__ __forceinline__ unsigned pkbf(float lo, float hi) {
    return __builtin_amdgcn_perm(__builtin_bit_cast(unsigned, hi),
                                 __builtin_bit_cast(unsigned, lo),
                                 0x07060302u);
}

// ---------------------------------------------------------------- utilities
__global__ void zero_kernel(float* p, long n) {
    long i = (long)blockIdx.x * blockDim.x + threadIdx.x;
    long stride = (long)gridDim.x * blockDim.x;
    for (; i < n; i += stride) p[i] = 0.0f;
}

// Wt[n][k] (bf16, k-contiguous): k<128 -> Wl[k][n], else Wr[k-128][n]
__global__ void wconv_kernel(const float* __restrict__ Wl,
                             const float* __restrict__ Wr,
                             unsigned short* __restrict__ wt) {
    int tid = blockIdx.x * blockDim.x + threadIdx.x;
    if (tid >= DIM * 2 * DIM) return;
    int n = tid >> 8;       // 0..127
    int k = tid & 255;      // 0..255
    float v = (k < DIM) ? Wl[k * DIM + n] : Wr[(k - DIM) * DIM + n];
    wt[n * 256 + k] = f2bfu(v);
}

__global__ void deg_kernel(const int* __restrict__ ei, float* __restrict__ deg) {
    int e = blockIdx.x * blockDim.x + threadIdx.x;
    if (e < N_EDGES) atomicAdd(&deg[ei[N_EDGES + e]], 1.0f);
}

__global__ void invd_kernel(float* deg) {
    int i = blockIdx.x * blockDim.x + threadIdx.x;
    if (i < N_NODES) deg[i] = 1.0f / fmaxf(deg[i], 1.0f);
}

// scatter-add h[src] rows into agg[dst]; 32 lanes x float4 per edge
__global__ __launch_bounds__(256)
void scatter_kernel(const int* __restrict__ ei, const float* __restrict__ h,
                    float* __restrict__ agg) {
    long tid = (long)blockIdx.x * blockDim.x + threadIdx.x;
    if (tid >= (long)N_EDGES * 32) return;
    int e = (int)(tid >> 5);
    int f = ((int)tid & 31) * 4;
    int s = ei[e];
    int d = ei[N_EDGES + e];
    const float4 v = *(const float4*)(h + (long)s * DIM + f);
    float* a = agg + (long)d * DIM + f;
    atomicAdd(a + 0, v.x);
    atomicAdd(a + 1, v.y);
    atomicAdd(a + 2, v.z);
    atomicAdd(a + 3, v.w);
}

// ---------------------------------------------------------------- fused SAGE GEMM
// out[r][c] = relu( (agg[r]*invd[r] ++ hsrc[r]) @ Wt^T + bias[c] )
// One wave = one 16x16 tile. block = (32 lanes, 8 col-tiles), grid.x = N/16.
__global__ __launch_bounds__(256)
void sage_gemm_kernel(const float* __restrict__ agg, const float* __restrict__ hsrc,
                      const float* __restrict__ invd, const unsigned short* __restrict__ wt,
                      const float* __restrict__ bias, float* __restrict__ out) {
    const int lane = threadIdx.x & 31;
    const int half = (lane >> 4) & 1;     // bounded 0/1 (helps range analysis)
    const int idx  = lane & 15;
    const int rowbase = blockIdx.x * 16;
    const int r = rowbase + idx;                  // A row held by this lane
    const int c = (threadIdx.y & 7) * 16 + idx;   // B col held by this lane
    const float scale = invd[r];

    // lane-private base pointers (include the half-wave K offset of 8)
    const float*          aggr = agg  + (long)r * DIM + half * 8;
    const float*          hr   = hsrc + (long)r * DIM + half * 8;
    const unsigned short* wc   = wt   + c * 256 + half * 8;

    v8f acc = {};

#pragma unroll
    for (int chunk = 0; chunk < 256; chunk += 32) {
        // branch depends ONLY on the unrolled induction var -> folds at compile time
        const bool isAgg = (chunk < DIM);
        const float* ab = isAgg ? (aggr + chunk) : (hr + (chunk - DIM));

        // A: lane's 16 elements = two contiguous 8-float runs (K pattern half*8+{0..7},{16..23})
        float4 a0 = *(const float4*)(ab + 0);
        float4 a1 = *(const float4*)(ab + 4);
        float4 a2 = *(const float4*)(ab + 16);
        float4 a3 = *(const float4*)(ab + 20);
        if (isAgg) {
            a0.x *= scale; a0.y *= scale; a0.z *= scale; a0.w *= scale;
            a1.x *= scale; a1.y *= scale; a1.z *= scale; a1.w *= scale;
            a2.x *= scale; a2.y *= scale; a2.z *= scale; a2.w *= scale;
            a3.x *= scale; a3.y *= scale; a3.z *= scale; a3.w *= scale;
        }

        // B: same K pattern, pre-packed bf16 -> two b128 loads
        uint4 b0 = *(const uint4*)(wc + chunk);        // dwords for vgprs 0..3
        uint4 b1 = *(const uint4*)(wc + chunk + 16);   // dwords for vgprs 4..7

        v8u ap, bp;
        ap[0] = pkbf(a0.x, a0.y);  ap[1] = pkbf(a0.z, a0.w);
        ap[2] = pkbf(a1.x, a1.y);  ap[3] = pkbf(a1.z, a1.w);
        ap[4] = pkbf(a2.x, a2.y);  ap[5] = pkbf(a2.z, a2.w);
        ap[6] = pkbf(a3.x, a3.y);  ap[7] = pkbf(a3.z, a3.w);
        bp[0] = b0.x; bp[1] = b0.y; bp[2] = b0.z; bp[3] = b0.w;
        bp[4] = b1.x; bp[5] = b1.y; bp[6] = b1.z; bp[7] = b1.w;

        acc = __builtin_amdgcn_wmma_f32_16x16x32_bf16(
            false, __builtin_bit_cast(v16bf, ap),
            false, __builtin_bit_cast(v16bf, bp),
            (short)0, acc, false, false);
    }

    const float b = bias[c];
#pragma unroll
    for (int i = 0; i < 8; ++i) {
        // C/D layout: vgpr i, lanes 0-15 -> M=i, lanes 16-31 -> M=i+8; N = idx
        int row = rowbase + half * 8 + i;
        out[(long)row * DIM + c] = fmaxf(acc[i] + b, 0.0f);
    }
}

// ---------------------------------------------------------------- 2-class head
__global__ void head_kernel(const float* __restrict__ h2, const float* __restrict__ Wlin,
                            const float* __restrict__ blin, float* __restrict__ out) {
    int n = blockIdx.x * blockDim.x + threadIdx.x;
    if (n >= N_NODES) return;
    float a0 = blin[0], a1 = blin[1];
    const float* hr = h2 + (long)n * DIM;
#pragma unroll 8
    for (int k = 0; k < DIM; ++k) {
        float h = hr[k];
        a0 += h * Wlin[k * 2 + 0];
        a1 += h * Wlin[k * 2 + 1];
    }
    out[n * 2 + 0] = a0;
    out[n * 2 + 1] = a1;
}

// ---------------------------------------------------------------- launcher
extern "C" void kernel_launch(void* const* d_in, const int* in_sizes, int n_in,
                              void* d_out, int out_size, void* d_ws, size_t ws_size,
                              hipStream_t stream) {
    const float* x    = (const float*)d_in[0];
    const int*   ei   = (const int*)  d_in[1];
    const float* W1l  = (const float*)d_in[2];
    const float* W1r  = (const float*)d_in[3];
    const float* b1   = (const float*)d_in[4];
    const float* W2l  = (const float*)d_in[5];
    const float* W2r  = (const float*)d_in[6];
    const float* b2   = (const float*)d_in[7];
    const float* Wlin = (const float*)d_in[8];
    const float* blin = (const float*)d_in[9];
    float* out = (float*)d_out;

    // workspace layout
    float* deg = (float*)d_ws;                               // N
    float* agg = deg + N_NODES;                              // N*128 (reused both layers)
    float* h1  = agg + (long)N_NODES * DIM;                  // N*128
    float* h2  = h1  + (long)N_NODES * DIM;                  // N*128
    unsigned short* wt1 = (unsigned short*)(h2 + (long)N_NODES * DIM);  // 128*256 bf16
    unsigned short* wt2 = wt1 + 256 * DIM;

    const int wconvN = DIM * 2 * DIM;
    const int scatterBlocks = (int)(((long)N_EDGES * 32 + 255) / 256);

    // zero deg + agg (contiguous)
    zero_kernel<<<2048, 256, 0, stream>>>(deg, (long)N_NODES * (DIM + 1));
    // weights -> transposed bf16
    wconv_kernel<<<(wconvN + 255) / 256, 256, 0, stream>>>(W1l, W1r, wt1);
    wconv_kernel<<<(wconvN + 255) / 256, 256, 0, stream>>>(W2l, W2r, wt2);
    // degrees, then invert in place
    deg_kernel<<<(N_EDGES + 255) / 256, 256, 0, stream>>>(ei, deg);
    invd_kernel<<<(N_NODES + 255) / 256, 256, 0, stream>>>(deg);
    // layer 1
    scatter_kernel<<<scatterBlocks, 256, 0, stream>>>(ei, x, agg);
    sage_gemm_kernel<<<dim3(N_NODES / 16), dim3(32, 8), 0, stream>>>(agg, x, deg, wt1, b1, h1);
    // layer 2
    zero_kernel<<<2048, 256, 0, stream>>>(agg, (long)N_NODES * DIM);
    scatter_kernel<<<scatterBlocks, 256, 0, stream>>>(ei, h1, agg);
    sage_gemm_kernel<<<dim3(N_NODES / 16), dim3(32, 8), 0, stream>>>(agg, h1, deg, wt2, b2, h2);
    // classifier head
    head_kernel<<<(N_NODES + 255) / 256, 256, 0, stream>>>(h2, Wlin, blin, out);
}
